// Net_19138374271116
// MI455X (gfx1250) — matrix operand order repaired and checked by reading, not compiled, not run on hardware
//
#include <hip/hip_runtime.h>
#include <hip/hip_bf16.h>
#include <cstdint>
#include <cstddef>

typedef __attribute__((ext_vector_type(16))) _Float16 v16h;
typedef __attribute__((ext_vector_type(8)))  float    v8f;

#define EPS_BN 1e-5f
#define KNB 64

// ---------------- WMMA fragment helpers ----------------
// A: 16x32 f16 from LDS/global, row-major with leading dim LD (elements).
// lanes0-15: row M=lane, half h=lane>>4: elems 0..7 -> K=8h+e, 8..15 -> K=16+8h+e
__device__ inline v16h frag_a(const _Float16* base, int ld, int row0, int k0, int lane) {
  int r = row0 + (lane & 15);
  int h = (lane >> 4) & 1;
  const _Float16* p = base + (size_t)r * ld + k0;
  v16h out;
#pragma unroll
  for (int e = 0; e < 8; ++e) {
    out[e]     = p[8 * h + e];
    out[e + 8] = p[16 + 8 * h + e];
  }
  return out;
}

// B: packed fragment layout. Weights pre-packed per (32K x 16N) tile as
// 512 contiguous f16: [lane][e], lane = (khalf<<4)|ncol, e -> K = 8*khalf+e (e<8), 16+8*khalf+(e-8) (e>=8).
// One aligned 32-byte load per lane -> 2x global_load_b128.
template <int NP>
__device__ inline v16h frag_bp(const _Float16* __restrict__ wp, int kc, int ct, int lane) {
  const _Float16* p = wp + (((size_t)(kc * (NP >> 4) + ct)) << 9) + (lane << 4);
  return *(const v16h*)p;
}

// ---------------- layer pointer bundle ----------------
struct LayerPtrs {
  const _Float16* W;   // packed fragment layout, zero padded
  const float* b;      // [Cout] bias
  const float* s;      // BN scale
  const float* t;      // BN shift
};
struct Mlp3P { LayerPtrs L[3]; };

// One MLP layer: cur[64 x KP] (f16, LDS) x W[KP x NP] -> nxt[64 x NP] (f16, LDS)
// with fused bias + ReLU + folded BN. Cout == NP for every layer in this net.
template <int KP, int NP>
__device__ inline void mlp_layer(const _Float16* __restrict__ cur, _Float16* __restrict__ nxt,
                                 LayerPtrs L, int wave, int lane) {
  constexpr int colTiles = NP >> 4;
  constexpr int nt = 4 * colTiles;   // 4 row tiles (64 rows) x colTiles
  for (int ti = wave; ti < nt; ti += 4) {
    int rt = ti / colTiles, ct = ti - rt * colTiles;
    int row0 = rt << 4, n0 = ct << 4;
    v8f acc = {};
#pragma unroll
    for (int kc = 0; kc < (KP >> 5); ++kc) {
      v16h a = frag_a(cur, KP, row0, kc << 5, lane);
      v16h w = frag_bp<NP>(L.W, kc, ct, lane);
      acc = __builtin_amdgcn_wmma_f32_16x16x32_f16(false, a, false, w,
                                                   (short)0, acc, false, false);
    }
    int col = n0 + (lane & 15);
    float bb = L.b[col], ss = L.s[col], tt = L.t[col];
    int rbase = row0 + ((lane >> 4) << 3);
#pragma unroll
    for (int v = 0; v < 8; ++v) {
      float h = fmaxf(acc[v] + bb, 0.f) * ss + tt;
      nxt[(rbase + v) * NP + col] = (_Float16)h;
    }
  }
}

// ---------------- prep kernels ----------------
// Pack f32 weights [Kin x Nout] into padded f16 fragment layout [Kp/32][Np/16][512].
__global__ void wpack_kernel(const float* __restrict__ src, _Float16* __restrict__ dst,
                             int Kin, int Nout, int Kp, int Np) {
  int i = blockIdx.x * blockDim.x + threadIdx.x;
  if (i >= Kp * Np) return;
  int tile = i >> 9;
  int rem = i & 511;
  int lane = rem >> 4, e = rem & 15;
  int tilesN = Np >> 4;
  int kt = tile / tilesN, nt = tile - kt * tilesN;
  int k = kt * 32 + ((e < 8) ? (((lane >> 4) << 3) + e) : (16 + ((lane >> 4) << 3) + (e - 8)));
  int n = nt * 16 + (lane & 15);
  float v = (k < Kin && n < Nout) ? src[(size_t)k * Nout + n] : 0.f;
  dst[i] = (_Float16)v;
}

__global__ void bnfold_kernel(const float* __restrict__ gamma, const float* __restrict__ beta,
                              const float* __restrict__ mean, const float* __restrict__ var,
                              float* __restrict__ s, float* __restrict__ t, int C) {
  int i = blockIdx.x * blockDim.x + threadIdx.x;
  if (i >= C) return;
  float sv = gamma[i] * rsqrtf(var[i] + EPS_BN);
  s[i] = sv;
  t[i] = beta[i] - mean[i] * sv;
}

// ---------------- farthest point sampling (one block per batch) ----------------
__global__ __launch_bounds__(256)
void fps_kernel(const float* __restrict__ pos, int* __restrict__ sel, int N, int m) {
  __shared__ float d2s[512];
  __shared__ float redv[256];
  __shared__ int   redi[256];
  __shared__ float lastp[3];
  __shared__ int   lastidx;
  int b = blockIdx.x;
  const float* p = pos + (size_t)b * N * 3;
  int tid = threadIdx.x;
  for (int i = tid; i < N; i += 256) d2s[i] = 1e10f;
  if (tid == 0) { lastidx = 0; sel[(size_t)b * m] = 0; }
  __syncthreads();
  for (int it = 1; it < m; ++it) {
    if (tid < 3) lastp[tid] = p[lastidx * 3 + tid];
    __syncthreads();
    float lx = lastp[0], ly = lastp[1], lz = lastp[2];
    float bv = -1.f; int bi = 0;
    for (int i = tid; i < N; i += 256) {
      float dx = p[i * 3] - lx, dy = p[i * 3 + 1] - ly, dz = p[i * 3 + 2] - lz;
      float nd = fminf(d2s[i], dx * dx + dy * dy + dz * dz);
      d2s[i] = nd;
      if (nd > bv) { bv = nd; bi = i; }
    }
    redv[tid] = bv; redi[tid] = bi;
    __syncthreads();
    for (int sft = 128; sft > 0; sft >>= 1) {
      if (tid < sft && redv[tid + sft] > redv[tid]) {
        redv[tid] = redv[tid + sft]; redi[tid] = redi[tid + sft];
      }
      __syncthreads();
    }
    if (tid == 0) { lastidx = redi[0]; sel[(size_t)b * m + it] = lastidx; }
    __syncthreads();
  }
}

// ---------------- gather query positions ----------------
__global__ void gather_pos_kernel(const float* __restrict__ pos, const int* __restrict__ sel,
                                  float* __restrict__ pos_q, int N, int M, int total) {
  int i = blockIdx.x * blockDim.x + threadIdx.x;
  if (i >= total) return;
  int b = i / (M * 3); int rem = i - b * M * 3; int qi = rem / 3; int d = rem - qi * 3;
  pos_q[i] = pos[((size_t)b * N + sel[(size_t)b * M + qi]) * 3 + d];
}

// ---------------- radius/top-k: one wave per query, 64 rounds of min extraction ----------------
__global__ __launch_bounds__(128)
void radius_kernel(const float* __restrict__ pos_src, const float* __restrict__ pos_q,
                   int* __restrict__ nbr, int N, int M, float r2) {
  __shared__ unsigned taken[4][16];
  int wv = threadIdx.x >> 5, lane = threadIdx.x & 31;
  int q = blockIdx.x * 4 + wv;
  int b = q / M;
  const float* ps = pos_src + (size_t)b * N * 3;
  const float* pq = pos_q + (size_t)q * 3;
  float qx = pq[0], qy = pq[1], qz = pq[2];
  if (lane < 16) taken[wv][lane] = 0u;
  __syncthreads();
  for (int round = 0; round < KNB; ++round) {
    float best = 3.0e38f; int bidx = -1;
    for (int j = lane; j < N; j += 32) {
      if (taken[wv][j >> 5] & (1u << (j & 31))) continue;
      float dx = ps[j * 3] - qx, dy = ps[j * 3 + 1] - qy, dz = ps[j * 3 + 2] - qz;
      float d2 = dx * dx + dy * dy + dz * dz;
      if (d2 <= r2 && d2 < best) { best = d2; bidx = j; }
    }
#pragma unroll
    for (int o = 16; o > 0; o >>= 1) {
      float ob = __shfl_xor(best, o, 32);
      int   oi = __shfl_xor(bidx, o, 32);
      if (oi >= 0 && (bidx < 0 || ob < best || (ob == best && oi < bidx))) { best = ob; bidx = oi; }
    }
    if (lane == 0) {
      nbr[(size_t)q * KNB + round] = bidx;
      if (bidx >= 0) taken[wv][bidx >> 5] |= (1u << (bidx & 31));
    }
    __syncthreads();
  }
}

// ---------------- fused SA module: gather -> 3-layer WMMA MLP -> masked max over K ----------------
// Compile-time dims: C input channels, K0 padded Cin (C+3 padded to 32),
// N0/N1/N2 layer output widths (all multiples of 16, Cout==Np).
template <int C, int K0, int N0, int N1, int N2>
__global__ __launch_bounds__(128)
void sa_fused_kernel(const float* __restrict__ xsrc, const float* __restrict__ pos_src,
                     const float* __restrict__ pos_q, const int* __restrict__ nbr,
                     float* __restrict__ xout, int N, int M, Mlp3P mlp) {
  constexpr int LDA = (K0 > N1) ? K0 : N1;
  constexpr int LDB = (N0 > N2) ? N0 : N2;
  __shared__ _Float16 bufA[KNB * LDA];
  __shared__ _Float16 bufB[KNB * LDB];
  __shared__ int rowvalid[KNB];
  __shared__ float qp[3];

  int q = blockIdx.x;
  int b = q / M;
  int tid = threadIdx.x;
  const float* xs = xsrc + (size_t)b * N * C;
  const float* ps = pos_src + (size_t)b * N * 3;
  const int*   nb = nbr + (size_t)q * KNB;
  const float* pq = pos_q + (size_t)q * 3;

  if (tid < 3) qp[tid] = pq[tid];
  if (tid < KNB) rowvalid[tid] = (nb[tid] >= 0) ? 1 : 0;
  // warm L2 path for weights while we gather features
  __builtin_prefetch(mlp.L[0].W, 0, 1);
  __builtin_prefetch(mlp.L[1].W, 0, 1);
  __builtin_prefetch(mlp.L[2].W, 0, 1);
  __syncthreads();

  constexpr int Cin = C + 3;
  for (int e = tid; e < KNB * K0; e += 128) {
    int row = e / K0, col = e - row * K0;
    int src = nb[row];
    float v = 0.f;
    if (src >= 0) {
      if (col < C)        v = xs[(size_t)src * C + col];
      else if (col < Cin) v = ps[src * 3 + (col - C)] - qp[col - C];
    }
    bufA[row * K0 + col] = (_Float16)v;
  }
  __syncthreads();

  int wave = tid >> 5, lane = tid & 31;
  mlp_layer<K0, N0>(bufA, bufB, mlp.L[0], wave, lane);
  __syncthreads();
  mlp_layer<N0, N1>(bufB, bufA, mlp.L[1], wave, lane);
  __syncthreads();
  mlp_layer<N1, N2>(bufA, bufB, mlp.L[2], wave, lane);
  __syncthreads();

  for (int c = tid; c < N2; c += 128) {
    float m = -3.0e38f;
#pragma unroll 4
    for (int r = 0; r < KNB; ++r)
      if (rowvalid[r]) m = fmaxf(m, (float)bufB[r * N2 + c]);
    xout[(size_t)q * N2 + c] = m;
  }
}

// ---------------- templated WMMA GEMM (+bias/relu/BN), one wave per 16x16 tile ----------------
template <int KP, int NP>
__global__ __launch_bounds__(32)
void gemm_bn_relu_kernel(const _Float16* __restrict__ A, const _Float16* __restrict__ Wp,
                         const float* __restrict__ bias, const float* __restrict__ s,
                         const float* __restrict__ t,
                         _Float16* __restrict__ out16, float* __restrict__ out32) {
  constexpr int tilesN = NP >> 4;
  int bt = blockIdx.x;
  int rt = bt / tilesN, ct = bt - rt * tilesN;
  int row0 = rt << 4, n0 = ct << 4;
  int lane = threadIdx.x;
  v8f acc = {};
#pragma unroll
  for (int kc = 0; kc < (KP >> 5); ++kc) {
    v16h a = frag_a(A, KP, row0, kc << 5, lane);
    v16h w = frag_bp<NP>(Wp, kc, ct, lane);
    acc = __builtin_amdgcn_wmma_f32_16x16x32_f16(false, a, false, w,
                                                 (short)0, acc, false, false);
  }
  int col = n0 + (lane & 15);
  float bb = bias[col], ss = s[col], tt = t[col];
  int rbase = row0 + ((lane >> 4) << 3);
#pragma unroll
  for (int v = 0; v < 8; ++v) {
    float h = fmaxf(acc[v] + bb, 0.f) * ss + tt;
    if (out16) out16[(size_t)(rbase + v) * NP + col] = (_Float16)h;
    if (out32) out32[(size_t)(rbase + v) * NP + col] = h;
  }
}

// ---------------- small glue kernels ----------------
__global__ void build_a3_kernel(const float* __restrict__ x3, const float* __restrict__ pos3,
                                _Float16* __restrict__ a3, int total) {
  int i = blockIdx.x * blockDim.x + threadIdx.x;
  if (i >= total) return;
  int row = i / 288, col = i - row * 288;
  float v = 0.f;
  if (col < 256)      v = x3[(size_t)row * 256 + col];
  else if (col < 259) v = pos3[(size_t)row * 3 + (col - 256)];
  a3[i] = (_Float16)v;
}

__global__ void maxpool_kernel(const float* __restrict__ h, float* __restrict__ pool, int total) {
  int i = blockIdx.x * blockDim.x + threadIdx.x;
  if (i >= total) return;
  int b = i >> 10, c = i & 1023;
  float m = -3.0e38f;
  for (int r = 0; r < 32; ++r)
    m = fmaxf(m, h[((size_t)(b * 32 + r) << 10) + c]);
  pool[i] = m;
}

__global__ void concat_head_kernel(const float* __restrict__ pool, const float* __restrict__ yg,
                                   float* __restrict__ cat, int total) {
  int i = blockIdx.x * blockDim.x + threadIdx.x;
  if (i >= total) return;
  int b = i / 1032, c = i - b * 1032;
  cat[i] = (c < 1024) ? pool[(size_t)b * 1024 + c] : yg[(size_t)b * 8 + (c - 1024)];
}

__global__ void lin_kernel(const float* __restrict__ in, const float* __restrict__ W,
                           const float* __restrict__ b, float* __restrict__ out,
                           int Bn, int Kin, int Nout, int relu) {
  int i = blockIdx.x * blockDim.x + threadIdx.x;
  if (i >= Bn * Nout) return;
  int row = i / Nout, col = i - row * Nout;
  float acc = b[col];
  const float* ip = in + (size_t)row * Kin;
  for (int k = 0; k < Kin; ++k) acc = fmaf(ip[k], W[(size_t)k * Nout + col], acc);
  if (relu) acc = fmaxf(acc, 0.f);
  out[i] = acc;
}

// ---------------- host launcher ----------------
extern "C" void kernel_launch(void* const* d_in, const int* in_sizes, int n_in,
                              void* d_out, int out_size, void* d_ws, size_t ws_size,
                              hipStream_t stream) {
  (void)in_sizes; (void)n_in; (void)out_size; (void)ws_size;
  const int B = 32;
  const float* x0   = (const float*)d_in[0];   // [B,512,128]
  const float* pos0 = (const float*)d_in[1];   // [B,512,3]
  const float* yg   = (const float*)d_in[2];   // [B,8]

  // 12 MLP layers: sa1(3), sa1a(3), sa2(3), sa3(3). Input index base = 3 + 6*l.
  const int NLAY = 12;
  const int kin_[NLAY]  = {131,128,128, 163,160,160, 195,192,192, 259,256,512};
  const int cout_[NLAY] = {128,128,160, 160,160,192, 192,192,256, 256,512,1024};
  const int kp_[NLAY]   = {160,128,128, 192,160,160, 224,192,192, 288,256,512};

  char* ws = (char*)d_ws;
  size_t off = 0;
  auto alloc = [&](size_t bytes) -> char* {
    char* p = ws + off;
    off = (off + bytes + 255) & ~(size_t)255;
    return p;
  };

  _Float16* Wh[NLAY]; float* Sc[NLAY]; float* Sh[NLAY];
  for (int l = 0; l < NLAY; ++l) {
    Wh[l] = (_Float16*)alloc((size_t)kp_[l] * cout_[l] * sizeof(_Float16));
    Sc[l] = (float*)alloc((size_t)cout_[l] * sizeof(float));
    Sh[l] = (float*)alloc((size_t)cout_[l] * sizeof(float));
  }
  int*   sel1 = (int*)alloc((size_t)B * 256 * 4);
  int*   sel2 = (int*)alloc((size_t)B * 128 * 4);
  int*   sel3 = (int*)alloc((size_t)B * 32 * 4);
  float* pos1 = (float*)alloc((size_t)B * 256 * 3 * 4);
  float* pos2 = (float*)alloc((size_t)B * 128 * 3 * 4);
  float* pos3 = (float*)alloc((size_t)B * 32 * 3 * 4);
  int*   nbr1 = (int*)alloc((size_t)B * 256 * KNB * 4);
  int*   nbr2 = (int*)alloc((size_t)B * 128 * KNB * 4);
  int*   nbr3 = (int*)alloc((size_t)B * 32 * KNB * 4);
  float* x1   = (float*)alloc((size_t)B * 256 * 160 * 4);
  float* x2   = (float*)alloc((size_t)B * 128 * 192 * 4);
  float* x3   = (float*)alloc((size_t)B * 32 * 256 * 4);
  _Float16* a3  = (_Float16*)alloc((size_t)B * 32 * 288 * 2);
  _Float16* h3a = (_Float16*)alloc((size_t)B * 32 * 256 * 2);
  _Float16* h3b = (_Float16*)alloc((size_t)B * 32 * 512 * 2);
  float* h3c  = (float*)alloc((size_t)B * 32 * 1024 * 4);
  float* pool = (float*)alloc((size_t)B * 1024 * 4);
  float* cat1 = (float*)alloc((size_t)B * 1032 * 4);
  float* hh1  = (float*)alloc((size_t)B * 512 * 4);
  float* hh2  = (float*)alloc((size_t)B * 256 * 4);
  float* hh3  = (float*)alloc((size_t)B * 128 * 4);

  // ---- prep: pack+pad weights to f16 fragment layout, fold BN ----
  for (int l = 0; l < NLAY; ++l) {
    const float* Wsrc  = (const float*)d_in[3 + 6 * l + 0];
    const float* gamma = (const float*)d_in[3 + 6 * l + 2];
    const float* beta  = (const float*)d_in[3 + 6 * l + 3];
    const float* mean  = (const float*)d_in[3 + 6 * l + 4];
    const float* var   = (const float*)d_in[3 + 6 * l + 5];
    int tot = kp_[l] * cout_[l];
    wpack_kernel<<<(tot + 255) / 256, 256, 0, stream>>>(Wsrc, Wh[l], kin_[l], cout_[l], kp_[l], cout_[l]);
    bnfold_kernel<<<(cout_[l] + 255) / 256, 256, 0, stream>>>(gamma, beta, mean, var, Sc[l], Sh[l], cout_[l]);
  }

  auto mk3 = [&](int l0) -> Mlp3P {
    Mlp3P m;
    for (int j = 0; j < 3; ++j) {
      int l = l0 + j;
      m.L[j].W = Wh[l];
      m.L[j].b = (const float*)d_in[3 + 6 * l + 1];
      m.L[j].s = Sc[l];
      m.L[j].t = Sh[l];
    }
    return m;
  };

  // ---- sa1: N=512 -> M=256, r=0.2, C=128 -> 160 ----
  fps_kernel<<<B, 256, 0, stream>>>(pos0, sel1, 512, 256);
  gather_pos_kernel<<<(B * 256 * 3 + 255) / 256, 256, 0, stream>>>(pos0, sel1, pos1, 512, 256, B * 256 * 3);
  radius_kernel<<<B * 256 / 4, 128, 0, stream>>>(pos0, pos1, nbr1, 512, 256, 0.04f);
  sa_fused_kernel<128, 160, 128, 128, 160><<<B * 256, 128, 0, stream>>>(
      x0, pos0, pos1, nbr1, x1, 512, 256, mk3(0));

  // ---- sa1a: N=256 -> M=128, r=0.2, C=160 -> 192 ----
  fps_kernel<<<B, 256, 0, stream>>>(pos1, sel2, 256, 128);
  gather_pos_kernel<<<(B * 128 * 3 + 255) / 256, 256, 0, stream>>>(pos1, sel2, pos2, 256, 128, B * 128 * 3);
  radius_kernel<<<B * 128 / 4, 128, 0, stream>>>(pos1, pos2, nbr2, 256, 128, 0.04f);
  sa_fused_kernel<160, 192, 160, 160, 192><<<B * 128, 128, 0, stream>>>(
      x1, pos1, pos2, nbr2, x2, 256, 128, mk3(3));

  // ---- sa2: N=128 -> M=32, r=0.4, C=192 -> 256 ----
  fps_kernel<<<B, 256, 0, stream>>>(pos2, sel3, 128, 32);
  gather_pos_kernel<<<(B * 32 * 3 + 255) / 256, 256, 0, stream>>>(pos2, sel3, pos3, 128, 32, B * 32 * 3);
  radius_kernel<<<B * 32 / 4, 128, 0, stream>>>(pos2, pos3, nbr3, 128, 32, 0.16f);
  sa_fused_kernel<192, 224, 192, 192, 256><<<B * 32, 128, 0, stream>>>(
      x2, pos2, pos3, nbr3, x3, 128, 32, mk3(6));

  // ---- sa3 (global): [B*32, 259] -> 256 -> 512 -> 1024, max over 32 points ----
  build_a3_kernel<<<(B * 32 * 288 + 255) / 256, 256, 0, stream>>>(x3, pos3, a3, B * 32 * 288);
  gemm_bn_relu_kernel<288, 256><<<(1024 / 16) * (256 / 16), 32, 0, stream>>>(
      a3, Wh[9], (const float*)d_in[3 + 6 * 9 + 1], Sc[9], Sh[9], h3a, nullptr);
  gemm_bn_relu_kernel<256, 512><<<(1024 / 16) * (512 / 16), 32, 0, stream>>>(
      h3a, Wh[10], (const float*)d_in[3 + 6 * 10 + 1], Sc[10], Sh[10], h3b, nullptr);
  gemm_bn_relu_kernel<512, 1024><<<(1024 / 16) * (1024 / 16), 32, 0, stream>>>(
      h3b, Wh[11], (const float*)d_in[3 + 6 * 11 + 1], Sc[11], Sh[11], nullptr, h3c);
  maxpool_kernel<<<(B * 1024 + 255) / 256, 256, 0, stream>>>(h3c, pool, B * 1024);

  // ---- head ----
  concat_head_kernel<<<(B * 1032 + 255) / 256, 256, 0, stream>>>(pool, yg, cat1, B * 1032);
  lin_kernel<<<(B * 512 + 255) / 256, 256, 0, stream>>>(
      cat1, (const float*)d_in[75], (const float*)d_in[76], hh1, B, 1032, 512, 1);
  lin_kernel<<<(B * 256 + 255) / 256, 256, 0, stream>>>(
      hh1, (const float*)d_in[77], (const float*)d_in[78], hh2, B, 512, 256, 1);
  lin_kernel<<<(B * 128 + 255) / 256, 256, 0, stream>>>(
      hh2, (const float*)d_in[79], (const float*)d_in[80], hh3, B, 256, 128, 1);
  lin_kernel<<<(B * 1 + 255) / 256, 256, 0, stream>>>(
      hh3, (const float*)d_in[81], (const float*)d_in[82], (float*)d_out, B, 128, 1, 0);
}
